// ACE_4174708211800
// MI455X (gfx1250) — matrix operand order repaired and checked by reading, not compiled, not run on hardware
//
// SEAN/SPADE normalization block for MI455X (gfx1250, wave32, WMMA + TDM).
// Convolutions are implicit GEMMs through v_wmma_f32_16x16x32_f16 with
// double-buffered LDS staging; weight tiles are staged by the Tensor Data
// Mover (tensor_load_to_lds, TENSORcnt) when the builtin is available.
#include <hip/hip_runtime.h>
#include <math.h>
#include <stdint.h>

typedef _Float16 h16;
typedef __attribute__((ext_vector_type(16))) h16   v16h;
typedef __attribute__((ext_vector_type(8)))  float v8f;
typedef __attribute__((ext_vector_type(4)))  unsigned u32x4;
typedef __attribute__((ext_vector_type(8)))  int      i32x8;
typedef __attribute__((ext_vector_type(4)))  int      i32x4;

struct alignas(16) Q16 { unsigned u[4]; };
struct alignas(8)  Q8  { unsigned u[2]; };

union FragA { v16h v; unsigned u[8]; };
union FragC { v8f  v; float    f[8]; };
union H8    { h16  h[8]; Q16 q; Q8 d[2]; unsigned u[4]; };

static constexpr int B_ = 8, C_ = 128, H_ = 256, W_ = 256, J_ = 19, D_ = 64;

#if defined(__gfx1250__) && __has_builtin(__builtin_amdgcn_tensor_load_to_lds)
#define USE_TDM 1
#else
#define USE_TDM 0
#endif

// ---------------------------------------------------------------------------
// Weight repack to f16. K-order for modes 1/2 is tap-major: k = tap*CIN + ci
// (contiguous ci for a fixed 3x3 tap -> vector 16B im2col loads).
// Shared conv stays ci-major (k = ci*9 + tap), padded 171 -> 192.
// ---------------------------------------------------------------------------
__global__ void k_pack(const float* __restrict__ gw, const float* __restrict__ bw,
                       const float* __restrict__ sgw, const float* __restrict__ sbw,
                       const float* __restrict__ shw,
                       h16* __restrict__ wavg, h16* __restrict__ wspade,
                       h16* __restrict__ wshared) {
  int i = blockIdx.x * 256 + threadIdx.x;
  if (i < 256 * 576) {                       // gamma_avg (rows 0..127) ; beta_avg
    int row = i / 576, k = i - row * 576;
    int tap = k >> 6, ci = k & 63;
    const float* src = (row < 128) ? gw + (size_t)row * 576
                                   : bw + (size_t)(row - 128) * 576;
    wavg[i] = (h16)src[ci * 9 + tap];
    return;
  }
  i -= 256 * 576;
  if (i < 256 * 1152) {                      // spade gamma ; spade beta
    int row = i / 1152, k = i - row * 1152;
    int tap = k >> 7, ci = k & 127;
    const float* src = (row < 128) ? sgw + (size_t)row * 1152
                                   : sbw + (size_t)(row - 128) * 1152;
    wspade[i] = (h16)src[ci * 9 + tap];
    return;
  }
  i -= 256 * 1152;
  if (i < 128 * 192) {                       // spade_shared, K padded to 192
    int row = i / 192, k = i - row * 192;
    int ci = k / 9, tap = k - ci * 9;
    wshared[i] = (ci < J_) ? (h16)shw[(size_t)row * (J_ * 9) + ci * 9 + tap]
                           : (h16)0.f;
  }
}

// mu[b,j,:] = relu(fc_w[j] @ style[b,j] + fc_b[j])
__global__ void k_mu(const float* __restrict__ style, const float* __restrict__ fcw,
                     const float* __restrict__ fcb, float* __restrict__ mu) {
  int bj = blockIdx.x;
  int j = bj % J_, b = bj / J_;
  int d = threadIdx.x;
  const float* s = style + ((size_t)b * J_ + j) * D_;
  const float* w = fcw + ((size_t)j * D_ + d) * D_;
  float acc = fcb[j * D_ + d];
  for (int i = 0; i < D_; i++) acc += s[i] * w[i];
  mu[((size_t)b * J_ + j) * D_ + d] = fmaxf(acc, 0.f);
}

// per-(b,c) instance-norm stats of x + noise[b,w,h]*nvar[c]
__global__ void k_stats(const float* __restrict__ x, const float* __restrict__ noise,
                        const float* __restrict__ nvar, float* __restrict__ meanp,
                        float* __restrict__ rstdp) {
  int bc = blockIdx.x;
  int b = bc >> 7, c = bc & 127;
  float nv = nvar[c];
  const float* xp = x + (size_t)bc * (H_ * W_);
  const float* np = noise + (size_t)b * (W_ * H_);
  float s = 0.f, ss = 0.f;
  for (int pix = threadIdx.x; pix < H_ * W_; pix += 256) {
    int h = pix >> 8, w = pix & 255;
    float v = xp[pix] + np[w * H_ + h] * nv;
    s += v; ss += v * v;
  }
  __shared__ float r1[256], r2[256];
  r1[threadIdx.x] = s; r2[threadIdx.x] = ss;
  __syncthreads();
  for (int o = 128; o > 0; o >>= 1) {
    if (threadIdx.x < o) { r1[threadIdx.x] += r1[threadIdx.x + o];
                           r2[threadIdx.x] += r2[threadIdx.x + o]; }
    __syncthreads();
  }
  if (threadIdx.x == 0) {
    float m = r1[0] * (1.f / 65536.f);
    float var = r2[0] * (1.f / 65536.f) - m * m;
    meanp[bc] = m;
    rstdp[bc] = rsqrtf(var + 1e-5f);
  }
}

// middle[b,h,w,:] = mu[b, last covering class, :] (NHWC f16), zeros if uncovered
__global__ void k_middle(const float* __restrict__ segmap, const float* __restrict__ mu,
                         h16* __restrict__ middle) {
  int idx = blockIdx.x * 256 + threadIdx.x;   // b*H*W
  int b = idx >> 16, pix = idx & 65535;
  int h = pix >> 8, w = pix & 255;
  const float* sm = segmap + (size_t)b * J_ * (128 * 128) + (h >> 1) * 128 + (w >> 1);
  int jstar = -1;
  for (int j = J_ - 1; j >= 0; j--) {
    if (sm[(size_t)j * 16384] > 0.f) { jstar = j; break; }
  }
  h16* out = middle + (size_t)idx * D_;
  if (jstar < 0) {
    Q16 z; z.u[0] = z.u[1] = z.u[2] = z.u[3] = 0;
    #pragma unroll
    for (int i = 0; i < 8; i++) *(Q16*)(out + i * 8) = z;
  } else {
    const float* m = mu + ((size_t)b * J_ + jstar) * D_;
    #pragma unroll
    for (int i = 0; i < 8; i++) {
      H8 p;
      #pragma unroll
      for (int r = 0; r < 8; r++) p.h[r] = (h16)m[i * 8 + r];
      *(Q16*)(out + i * 8) = p.q;
    }
  }
}

// ---------------------------------------------------------------------------
// Implicit-GEMM 3x3 conv, WMMA f32_16x16x32_f16, double-buffered LDS.
// Block tile: 64 pixels (one row segment) x MTILE output channels.
// 8 waves: nsub = wave&3 (16-pixel subtile), mgroup = wave>>2 (M half).
// Weight slices staged by the TDM (tensor_load_to_lds) with LDS row padding
// 64B data + 8B pad -> 72B stride matching the bank-conflict-free layout.
// MODE 0: seg -> relu(conv) -> actv            (M=128, K=192)
// MODE 1: middle -> ga*gamma_avg, ba*beta_avg  (M=256, K=576)
// MODE 2: actv -> spade gamma/beta, fused blend + instance-norm epilogue
// ---------------------------------------------------------------------------
template <int MODE>
__global__ __launch_bounds__(256) void k_conv(
    const h16* __restrict__ wpk, const h16* __restrict__ inh,
    const float* __restrict__ segmap,
    const float* __restrict__ bias0, const float* __restrict__ bias1,
    const float* __restrict__ blendg, const float* __restrict__ blendb,
    h16* __restrict__ out0, h16* __restrict__ out1,
    const h16* __restrict__ gpart, const h16* __restrict__ bpart,
    const float* __restrict__ x, const float* __restrict__ noise,
    const float* __restrict__ nvar, const float* __restrict__ meanp,
    const float* __restrict__ rstdp, float* __restrict__ outf) {
  constexpr int MTILE = (MODE == 0) ? 128 : 256;
  constexpr int KFULL = (MODE == 0) ? 192 : ((MODE == 1) ? 576 : 1152);
  constexpr int CIN   = (MODE == 1) ? 64 : 128;
  constexpr int LSH   = (MODE == 1) ? 6 : 7;
  constexpr int NMS   = MTILE / 32;       // m-subtiles per wave
  constexpr int NST   = KFULL / 32;       // K steps
  constexpr int BUFH  = MTILE * 36 + 64 * 36;           // halves per buffer
  constexpr int SMEMB = (BUFH * 4 > 34816) ? BUFH * 4 : 34816;

  __shared__ __align__(16) char smem[SMEMB];
  h16* const base = (h16*)smem;

  const int tid = threadIdx.x;
  const int lane = tid & 31, wave = tid >> 5;
  const int nsub = wave & 3, mgroup = wave >> 2;
  const int hl = lane >> 4, l16 = lane & 15;

  const int b = blockIdx.z, h = blockIdx.y, w0 = blockIdx.x << 6;

  FragC acc[NMS];
  #pragma unroll
  for (int ms = 0; ms < NMS; ms++)
    #pragma unroll
    for (int r = 0; r < 8; r++) acc[ms].f[r] = 0.f;

  const int sp = tid >> 2;              // staging pixel 0..63
  const int skb = (tid & 3) << 3;       // staging k-offset {0,8,16,24}

  // ---- stage A: weight slice (MTILE x 32 K), TDM when available ----
  auto stageA = [&](int k0, h16* As) {
#if USE_TDM
    if (wave == 0) {
      unsigned long long ga =
          (unsigned long long)(uintptr_t)wpk + (unsigned long long)k0 * 2ull;
      unsigned lds_off = (unsigned)(uintptr_t)(void*)As;   // low 32b = LDS offset
      u32x4 g0;
      g0[0] = 1u;                                          // count=1, user D#
      g0[1] = lds_off;                                     // lds_addr
      g0[2] = (unsigned)ga;                                // global_addr[31:0]
      g0[3] = (unsigned)((ga >> 32) & 0x1ffffffu) | (2u << 30); // addr[56:32]|type=2
      i32x8 g1;
      // data_size=2B(1), pad_enable, pad_interval=16dw(3), pad_amount=2dw(1)
      g1[0] = (int)((1u << 16) | (1u << 20) | (3u << 22) | (1u << 25));
      g1[1] = (int)((unsigned)(KFULL & 0xffff) << 16);     // tensor_dim0[15:0]
      g1[2] = (int)((unsigned)MTILE << 16);                // tensor_dim1[15:0]
      g1[3] = (int)(32u << 16);                            // tile_dim0=32
      g1[4] = (int)MTILE;                                  // tile_dim1
      g1[5] = (int)KFULL;                                  // tensor_dim0_stride
      g1[6] = 0; g1[7] = 0;
      i32x4 g2; g2[0] = 0; g2[1] = 0; g2[2] = 0; g2[3] = 0;
      i32x4 g3 = g2;
      i32x8 g4;                                            // trailing group (6-arg form)
      g4[0] = 0; g4[1] = 0; g4[2] = 0; g4[3] = 0;
      g4[4] = 0; g4[5] = 0; g4[6] = 0; g4[7] = 0;
      __builtin_amdgcn_tensor_load_to_lds(g0, g1, g2, g3, g4, 0);
    }
#else
    if (tid < MTILE) {
      const Q8* src = (const Q8*)(wpk + (size_t)tid * KFULL + k0);
      Q8* dst = (Q8*)(As + tid * 36);
      #pragma unroll
      for (int i = 0; i < 8; i++) dst[i] = src[i];
    }
#endif
  };

  // ---- stage B: im2col patch tile (64 pixels x 32 K) ----
  auto stageB = [&](int k0, h16* Bs) {
    H8 pk;
    const int wq = w0 + sp;
    const int k = k0 + skb;
    if constexpr (MODE == 0) {
      #pragma unroll
      for (int i = 0; i < 8; i++) {
        int kk = k + i;
        int ci = kk / 9;
        int tap = kk - ci * 9;
        int kh = tap / 3, kw = tap - kh * 3;
        int hh = h + kh - 1, ww = wq + kw - 1;
        float v = 0.f;
        if (ci < J_ && (unsigned)hh < (unsigned)H_ && (unsigned)ww < (unsigned)W_)
          v = segmap[(((size_t)b * J_ + ci) * (H_ / 2) + (hh >> 1)) * (W_ / 2) + (ww >> 1)];
        pk.h[i] = (h16)v;
      }
    } else {
      int tap = k >> LSH;               // same tap across all 8 k's
      int ci = k & (CIN - 1);
      int kh = tap / 3, kw = tap - kh * 3;
      int hh = h + kh - 1, ww = wq + kw - 1;
      if ((unsigned)hh < (unsigned)H_ && (unsigned)ww < (unsigned)W_) {
        pk.q = *(const Q16*)(inh + (((size_t)b * H_ + hh) * W_ + ww) * CIN + ci);
      } else {
        pk.u[0] = pk.u[1] = pk.u[2] = pk.u[3] = 0;
      }
    }
    Q8* bs = (Q8*)(Bs + sp * 36 + skb);
    bs[0] = pk.d[0]; bs[1] = pk.d[1];
  };

  // ---- compute: 16x16x32 WMMA per m-subtile ----
  auto compute = [&](h16* As, h16* Bs) {
    FragA bf;   // B frag: lane = column N, vgpr<4 -> K = hl*8+2v ; vgpr>=4 -> K = 16+hl*8+2(v-4)
    const unsigned* bsrc = (const unsigned*)(Bs + (nsub * 16 + l16) * 36 + hl * 8);
    const unsigned* bsrc1 = (const unsigned*)(Bs + (nsub * 16 + l16) * 36 + 16 + hl * 8);
    #pragma unroll
    for (int v = 0; v < 4; v++) { bf.u[v] = bsrc[v]; bf.u[v + 4] = bsrc1[v]; }
    #pragma unroll
    for (int ms = 0; ms < NMS; ms++) {
      const h16* arow = As + ((mgroup * NMS + ms) * 16 + l16) * 36;
      FragA af; // A frag: vgpr<4 -> K = hl*8+2v ; vgpr>=4 -> K = 16+hl*8+2(v-4)
      const unsigned* a0 = (const unsigned*)(arow + hl * 8);
      const unsigned* a1 = (const unsigned*)(arow + 16 + hl * 8);
      #pragma unroll
      for (int v = 0; v < 4; v++) { af.u[v] = a0[v]; af.u[v + 4] = a1[v]; }
      acc[ms].v = __builtin_amdgcn_wmma_f32_16x16x32_f16(
          false, af.v, false, bf.v, (short)0, acc[ms].v, false, false);
    }
  };

  // ---- double-buffered pipeline: one barrier per K step ----
  stageA(0, base);
  stageB(0, base + MTILE * 36);
#if USE_TDM
  if (wave == 0) __builtin_amdgcn_s_wait_tensorcnt(0);
#endif
  __syncthreads();
  for (int i = 0; i < NST; i++) {
    h16* cur = base + (i & 1) * BUFH;
    h16* nxt = base + ((i + 1) & 1) * BUFH;
    if (i + 1 < NST) {
      stageA((i + 1) * 32, nxt);
      stageB((i + 1) * 32, nxt + MTILE * 36);
    }
    compute(cur, cur + MTILE * 36);
#if USE_TDM
    if (wave == 0 && i + 1 < NST) __builtin_amdgcn_s_wait_tensorcnt(0);
#endif
    __syncthreads();
  }

  const size_t pixb = ((size_t)b * H_ + h) * W_ + w0;
  const int p = nsub * 16 + l16;

  if constexpr (MODE == 0) {
    #pragma unroll
    for (int ms = 0; ms < NMS; ms++) {
      int co = (mgroup * NMS + ms) * 16 + hl * 8;
      H8 o;
      #pragma unroll
      for (int r = 0; r < 8; r++)
        o.h[r] = (h16)fmaxf(acc[ms].f[r] + bias0[co + r], 0.f);
      *(Q16*)(out0 + (pixb + p) * 128 + co) = o.q;
    }
  } else if constexpr (MODE == 1) {
    float ga = 1.f / (1.f + expf(-blendg[0]));
    float ba = 1.f / (1.f + expf(-blendb[0]));
    const float* bb = mgroup ? bias1 : bias0;
    h16* op = mgroup ? out1 : out0;
    float sc = mgroup ? ba : ga;
    #pragma unroll
    for (int ms = 0; ms < NMS; ms++) {
      int co = ms * 16 + hl * 8;        // mgroup offset of 128 cancels
      H8 o;
      #pragma unroll
      for (int r = 0; r < 8; r++)
        o.h[r] = (h16)(sc * (acc[ms].f[r] + bb[co + r]));
      *(Q16*)(op + (pixb + p) * 128 + co) = o.q;
    }
  } else {
    float ga = 1.f / (1.f + expf(-blendg[0]));
    float ba = 1.f / (1.f + expf(-blendb[0]));
    h16* Gs = (h16*)smem;               // [64][136] gamma_final (overlays buffers)
    h16* Bf = Gs + 64 * 136;            // [64][136] beta_final
    const h16* pin = mgroup ? bpart : gpart;
    const float* bb = mgroup ? bias1 : bias0;
    float om = mgroup ? (1.f - ba) : (1.f - ga);
    h16* tile = mgroup ? Bf : Gs;
    const size_t pix = pixb + p;
    #pragma unroll
    for (int ms = 0; ms < NMS; ms++) {
      int co = ms * 16 + hl * 8;
      H8 gp; gp.q = *(const Q16*)(pin + pix * 128 + co);
      H8 o;
      #pragma unroll
      for (int r = 0; r < 8; r++)
        o.h[r] = (h16)((float)gp.h[r] + om * (acc[ms].f[r] + bb[co + r]));
      *(Q16*)(tile + p * 136 + co) = o.q;
    }
    __syncthreads();
    // fused instance-norm + blend epilogue, coalesced over w
    for (int i = tid; i < 64 * 128; i += 256) {
      int p2 = i & 63, c = i >> 6;
      int wq = w0 + p2;
      float g  = (float)Gs[p2 * 136 + c];
      float be = (float)Bf[p2 * 136 + c];
      size_t xi = (((size_t)b * C_ + c) * H_ + h) * W_ + wq;
      float xn = x[xi] + noise[((size_t)b * W_ + wq) * H_ + h] * nvar[c];
      float nm = (xn - meanp[b * C_ + c]) * rstdp[b * C_ + c];
      outf[xi] = nm * (1.f + g) + be;
    }
  }
}

// ---------------------------------------------------------------------------
extern "C" void kernel_launch(void* const* d_in, const int* in_sizes, int n_in,
                              void* d_out, int out_size, void* d_ws, size_t ws_size,
                              hipStream_t stream) {
  const float* x       = (const float*)d_in[0];
  const float* segmap  = (const float*)d_in[1];
  const float* style   = (const float*)d_in[2];
  const float* noise   = (const float*)d_in[3];
  const float* nvar    = (const float*)d_in[4];
  const float* blendg  = (const float*)d_in[5];
  const float* blendb  = (const float*)d_in[6];
  const float* fcw     = (const float*)d_in[7];
  const float* fcb     = (const float*)d_in[8];
  const float* cgw     = (const float*)d_in[9];
  const float* cgb     = (const float*)d_in[10];
  const float* cbw     = (const float*)d_in[11];
  const float* cbb     = (const float*)d_in[12];
  const float* ssw     = (const float*)d_in[13];
  const float* ssb     = (const float*)d_in[14];
  const float* sgw     = (const float*)d_in[15];
  const float* sgb     = (const float*)d_in[16];
  const float* sbw     = (const float*)d_in[17];
  const float* sbb     = (const float*)d_in[18];
  float* out = (float*)d_out;

  char* ws = (char*)d_ws;
  size_t o = 0;
  float* mu    = (float*)(ws + o); o += (size_t)B_ * J_ * D_ * 4;        // 38912
  float* meanp = (float*)(ws + o); o += (size_t)B_ * C_ * 4;             // 4096
  float* rstdp = (float*)(ws + o); o += (size_t)B_ * C_ * 4;             // 4096
  h16* wavg    = (h16*)(ws + o);   o += (size_t)256 * 576 * 2;           // 294912
  h16* wspade  = (h16*)(ws + o);   o += (size_t)256 * 1152 * 2;          // 589824
  h16* wshared = (h16*)(ws + o);   o += (size_t)128 * 192 * 2;           // 49152
  h16* middle  = (h16*)(ws + o);   o += (size_t)B_ * H_ * W_ * 64 * 2;   // 64 MiB
  h16* actv    = (h16*)(ws + o);   o += (size_t)B_ * H_ * W_ * 128 * 2;  // 128 MiB
  h16* gpart   = (h16*)(ws + o);   o += (size_t)B_ * H_ * W_ * 128 * 2;  // 128 MiB
  h16* bpart   = (h16*)(ws + o);   o += (size_t)B_ * H_ * W_ * 128 * 2;  // 128 MiB

  k_pack<<<1824, 256, 0, stream>>>(cgw, cbw, sgw, sbw, ssw, wavg, wspade, wshared);
  k_mu<<<B_ * J_, 64, 0, stream>>>(style, fcw, fcb, mu);
  k_stats<<<B_ * C_, 256, 0, stream>>>(x, noise, nvar, meanp, rstdp);
  k_middle<<<(B_ * H_ * W_) / 256, 256, 0, stream>>>(segmap, mu, middle);

  dim3 grid(W_ / 64, H_, B_);
  // spade_shared -> actv
  k_conv<0><<<grid, 256, 0, stream>>>(wshared, nullptr, segmap, ssb, nullptr,
                                      nullptr, nullptr, actv, nullptr,
                                      nullptr, nullptr, nullptr, nullptr,
                                      nullptr, nullptr, nullptr, nullptr);
  // middle -> ga*gamma_avg, ba*beta_avg partials
  k_conv<1><<<grid, 256, 0, stream>>>(wavg, middle, nullptr, cgb, cbb,
                                      blendg, blendb, gpart, bpart,
                                      nullptr, nullptr, nullptr, nullptr,
                                      nullptr, nullptr, nullptr, nullptr);
  // actv -> spade gamma/beta, fused blend + instance-norm -> out
  k_conv<2><<<grid, 256, 0, stream>>>(wspade, actv, nullptr, sgb, sbb,
                                      blendg, blendb, nullptr, nullptr,
                                      gpart, bpart, x, noise, nvar,
                                      meanp, rstdp, out);
  (void)in_sizes; (void)n_in; (void)out_size; (void)ws_size;
}
